// Sentence_word_Attention_82068235092502
// MI455X (gfx1250) — matrix operand order repaired and checked by reading, not compiled, not run on hardware
//
#include <hip/hip_runtime.h>

namespace {

constexpr int kD  = 1024;   // d_model (K and N of every projection)
constexpr int kLw = 2048;
constexpr int kB  = 2;
constexpr int kM  = kB * kLw;  // 4096 token rows

typedef __attribute__((ext_vector_type(16))) __bf16 v16bf;
typedef __attribute__((ext_vector_type(8)))  __bf16 v8bf;
typedef __attribute__((ext_vector_type(8)))  float  v8f;
typedef __attribute__((ext_vector_type(8)))  unsigned short us8;
typedef __attribute__((ext_vector_type(4)))  float  f4;

#if defined(__HIP_DEVICE_COMPILE__) && \
    __has_builtin(__builtin_amdgcn_global_load_async_to_lds_b128)
#define USE_ASYNC_LDS 1
#else
#define USE_ASYNC_LDS 0
#endif

// Match the builtin's parameter types exactly:
//   (v4i addrspace(1)*, v4i addrspace(3)*, imm offset, imm cpol)
typedef int v4i __attribute__((__vector_size__(4 * sizeof(int))));
typedef __attribute__((address_space(1))) v4i* gv4ip;
typedef __attribute__((address_space(3))) v4i* lv4ip;

__device__ __forceinline__ unsigned short f2bf(float f) {
  union { float f; unsigned u; } a; a.f = f;
  unsigned r = a.u + 0x7FFFu + ((a.u >> 16) & 1u);   // round-to-nearest-even
  return (unsigned short)(r >> 16);
}
__device__ __forceinline__ __bf16 bfc(float f) {
  return __builtin_bit_cast(__bf16, f2bf(f));
}

// 16-byte global -> LDS copy; async DMA (ASYNCcnt) when available.
__device__ __forceinline__ void cp16(const unsigned short* g, unsigned short* l) {
#if USE_ASYNC_LDS
  __builtin_amdgcn_global_load_async_to_lds_b128((gv4ip)g, (lv4ip)l, 0, 0);
#else
  *(us8*)l = *(const us8*)g;
#endif
}
// Ensure this wave's staging writes have landed in LDS (before barrier).
__device__ __forceinline__ void stage_fence() {
#if USE_ASYNC_LDS
#if __has_builtin(__builtin_amdgcn_s_wait_asynccnt)
  __builtin_amdgcn_s_wait_asynccnt(0);
#else
  asm volatile("s_wait_asynccnt 0" ::: "memory");
#endif
#endif
  asm volatile("s_wait_dscnt 0" ::: "memory");
}

__device__ __forceinline__ v16bf join16(v8bf lo, v8bf hi) {
  v16bf a;
#pragma unroll
  for (int i = 0; i < 8; ++i) { a[i] = lo[i]; a[i + 8] = hi[i]; }
  return a;
}

// A-operand fragment of a 16x32 bf16 tile; lds row-major [row][k].
__device__ __forceinline__ v16bf frag_a(const unsigned short* lds, int row0,
                                        int k0, int stride, int lane) {
  const int m  = lane & 15;
  const int kb = (lane >> 4) << 3;
  const unsigned short* p = lds + (row0 + m) * stride + k0 + kb;
  return join16(*(const v8bf*)p, *(const v8bf*)(p + 16));
}
// B-operand fragment of a 32x16 bf16 tile; lds n-major [n][k].
__device__ __forceinline__ v16bf frag_b(const unsigned short* lds, int n0,
                                        int k0, int stride, int lane) {
  const int n  = lane & 15;
  const int kb = (lane >> 4) << 3;
  const unsigned short* p = lds + (n0 + n) * stride + k0 + kb;
  return join16(*(const v8bf*)p, *(const v8bf*)(p + 16));
}

__device__ __forceinline__ v8f wmma_bf16(v16bf a, v16bf b, v8f c) {
  return __builtin_amdgcn_wmma_f32_16x16x32_bf16(false, a, false, b,
                                                 (short)0, c, false, false);
}

// ---------------------------------------------------------------------------
// Elementwise f32 -> bf16 (one-time pre-conversion of word + weights).
// ---------------------------------------------------------------------------
__global__ __launch_bounds__(256)
void to_bf16_kernel(const float* __restrict__ x, unsigned short* __restrict__ y,
                    int n) {
  const int i = (blockIdx.x * 256 + threadIdx.x) * 8;
  if (i >= n) return;
  const f4 a = *(const f4*)&x[i];
  const f4 b = *(const f4*)&x[i + 4];
  us8 o;
#pragma unroll
  for (int j = 0; j < 4; ++j) { o[j] = f2bf(a[j]); o[4 + j] = f2bf(b[j]); }
  *(us8*)&y[i] = o;
}

// ---------------------------------------------------------------------------
// Y[kM,1024] = X @ W^T + bias, X/W pre-converted bf16. 64x128 workgroup tile,
// 8 waves x (2x2 WMMA tiles); double-buffered LDS, async staging, 1 barrier/K.
// ---------------------------------------------------------------------------
template <bool OUT_BF16>
__global__ __launch_bounds__(256)
void proj_gemm(const unsigned short* __restrict__ x,
               const unsigned short* __restrict__ w,
               const float* __restrict__ bias, void* __restrict__ yv) {
  __shared__ __align__(16) unsigned short ldsA[2][64 * 40];
  __shared__ __align__(16) unsigned short ldsB[2][128 * 40];

  const int tid = threadIdx.x, lane = tid & 31, wave = tid >> 5;
  const int wn = wave & 3, wm = wave >> 2;
  const int n0 = blockIdx.x * 128, m0 = blockIdx.y * 64;

  v8f acc[2][2] = {};

  const int ar = tid >> 2, ak = (tid & 3) << 3;   // A: 64 rows x 32 k
  const int bn = tid & 127, bk = (tid >> 7) << 4; // B: 128 n x 32 k

  auto stage = [&](int kk, int buf) {
    cp16(&x[(m0 + ar) * kD + kk + ak], &ldsA[buf][ar * 40 + ak]);
    cp16(&w[(n0 + bn) * kD + kk + bk], &ldsB[buf][bn * 40 + bk]);
    cp16(&w[(n0 + bn) * kD + kk + bk + 8], &ldsB[buf][bn * 40 + bk + 8]);
  };

  stage(0, 0);
  for (int it = 0; it < kD / 32; ++it) {
    const int cur = it & 1;
    stage_fence();
    __syncthreads();
    if (it + 1 < kD / 32) stage((it + 1) * 32, cur ^ 1);

    v16bf af[2], bfm[2];
#pragma unroll
    for (int mi = 0; mi < 2; ++mi)
      af[mi] = frag_a(ldsA[cur], wm * 32 + mi * 16, 0, 40, lane);
#pragma unroll
    for (int ni = 0; ni < 2; ++ni)
      bfm[ni] = frag_b(ldsB[cur], wn * 32 + ni * 16, 0, 40, lane);
#pragma unroll
    for (int mi = 0; mi < 2; ++mi)
#pragma unroll
      for (int ni = 0; ni < 2; ++ni)
        acc[mi][ni] = wmma_bf16(af[mi], bfm[ni], acc[mi][ni]);
  }

  const int nC = lane & 15;
  const int mC = (lane >> 4) << 3;
#pragma unroll
  for (int mi = 0; mi < 2; ++mi)
#pragma unroll
    for (int ni = 0; ni < 2; ++ni) {
      const int col = n0 + wn * 32 + ni * 16 + nC;
      const float bv = bias[col];
#pragma unroll
      for (int r = 0; r < 8; ++r) {
        const int row = m0 + wm * 32 + mi * 16 + mC + r;
        const float val = acc[mi][ni][r] + bv;
        if (OUT_BF16) ((unsigned short*)yv)[row * kD + col] = f2bf(val);
        else          ((float*)yv)[row * kD + col] = val;
      }
    }
}

// ---------------------------------------------------------------------------
// Flash attention, one workgroup per (b,h,64-query block); 4 waves x 16 rows.
// S is transposed through per-wave LDS into A-layout so softmax reductions are
// in-lane; P is assembled directly into WMMA A-fragments. K/V double-buffered.
// ---------------------------------------------------------------------------
__global__ __launch_bounds__(128)
void attn_kernel(const unsigned short* __restrict__ q,
                 const unsigned short* __restrict__ k,
                 const unsigned short* __restrict__ v,
                 unsigned short* __restrict__ ctx) {
  __shared__ __align__(16) unsigned short ldsQ[64 * 72];       // [qrow][d]
  __shared__ __align__(16) unsigned short ldsK[2][32 * 72];    // [j][d]
  __shared__ __align__(16) unsigned short ldsVT[2][64 * 40];   // [d][j]
  __shared__ __align__(16) float          ldsS[4 * 16 * 36];   // per-wave S

  const int tid  = threadIdx.x, lane = tid & 31, wave = tid >> 5;
  const int qb0  = blockIdx.x * 64;
  const int bh   = blockIdx.y;
  const int b    = bh >> 4;
  const int h    = bh & 15;
  const int rowbase = b * kLw;
  const int hcol    = h * 64;

  // ---- stage Q (64 x 64 bf16) ----
  {
    const int row = tid >> 1, dh = (tid & 1) * 32;
    const unsigned short* src = q + ((rowbase + qb0 + row) * kD + hcol + dh);
#pragma unroll
    for (int i = 0; i < 4; ++i)
      cp16(&src[i * 8], &ldsQ[row * 72 + dh + i * 8]);
  }

  const int kjr = tid >> 2, kjd = (tid & 3) << 4;   // K: 32 j x 64 d
  const int vj  = tid & 31, vd = (tid >> 5) << 4;   // V: 32 j x 64 d

  auto stageK = [&](int jb, int buf) {
    const unsigned short* src = k + ((rowbase + jb * 32 + kjr) * kD + hcol + kjd);
    cp16(&src[0], &ldsK[buf][kjr * 72 + kjd]);
    cp16(&src[8], &ldsK[buf][kjr * 72 + kjd + 8]);
  };
  auto stageV = [&](int jb, int buf) {   // transpose into [d][j]
    const unsigned short* src = v + ((rowbase + jb * 32 + vj) * kD + hcol + vd);
    const us8 v0 = *(const us8*)&src[0];
    const us8 v1 = *(const us8*)&src[8];
#pragma unroll
    for (int i = 0; i < 8; ++i) {
      ldsVT[buf][(vd + i) * 40 + vj]     = v0[i];
      ldsVT[buf][(vd + 8 + i) * 40 + vj] = v1[i];
    }
  };

  stageK(0, 0);
  stageV(0, 0);
  stage_fence();
  __syncthreads();

  const v16bf a0 = frag_a(ldsQ, wave * 16, 0,  72, lane);  // d 0..31
  const v16bf a1 = frag_a(ldsQ, wave * 16, 32, 72, lane);  // d 32..63

  v8f o[4] = {};
  float mrow = -3.0e38f, lrow = 0.0f;  // per-lane row state (A-layout rows)
  const int nC = lane & 15, mC = (lane >> 4) << 3;
  float* myS = &ldsS[wave * 16 * 36];
  const int sr  = lane & 15;           // S reload row (A layout)
  const int skb = (lane >> 4) << 3;    // S reload col base {0,8}

  for (int jb = 0; jb < kLw / 32; ++jb) {
    const int cur = jb & 1;
    if (jb) { stage_fence(); __syncthreads(); }
    if (jb + 1 < kLw / 32) { stageK(jb + 1, cur ^ 1); stageV(jb + 1, cur ^ 1); }

    // S = Q K^T (16 x 32 tile per wave; k-dim = 64 -> 2 steps per col tile)
    v8f s0 = {}, s1 = {};
    s0 = wmma_bf16(a0, frag_b(ldsK[cur], 0, 0, 72, lane), s0);
    s0 = wmma_bf16(a1, frag_b(ldsK[cur], 0, 32, 72, lane), s0);
    s1 = wmma_bf16(a0, frag_b(ldsK[cur], 16, 0, 72, lane), s1);
    s1 = wmma_bf16(a1, frag_b(ldsK[cur], 16, 32, 72, lane), s1);

    // C-layout -> LDS (f32), reload in A-layout: each lane owns half a row
#pragma unroll
    for (int r = 0; r < 8; ++r) {
      myS[(mC + r) * 36 + nC]      = s0[r];
      myS[(mC + r) * 36 + 16 + nC] = s1[r];
    }
    asm volatile("s_wait_dscnt 0" ::: "memory");
    const float* sp = &myS[sr * 36 + skb];
    const f4 c0 = *(const f4*)sp;          // cols skb+0..3
    const f4 c1 = *(const f4*)(sp + 4);    // cols skb+4..7
    const f4 c2 = *(const f4*)(sp + 16);   // cols 16+skb+0..3
    const f4 c3 = *(const f4*)(sp + 20);   // cols 16+skb+4..7

    float rmax = c0[0];
#pragma unroll
    for (int i = 1; i < 4; ++i) rmax = fmaxf(rmax, c0[i]);
#pragma unroll
    for (int i = 0; i < 4; ++i) rmax = fmaxf(rmax, fmaxf(c1[i], fmaxf(c2[i], c3[i])));
    rmax = fmaxf(rmax, __shfl_xor(rmax, 16, 32));   // join the two row halves

    const float mnew = fmaxf(mrow, rmax);
    const float corr = __expf(mrow - mnew);
    mrow = mnew;

    v16bf pa;            // P directly in WMMA A-fragment order
    float rs = 0.0f;
#pragma unroll
    for (int i = 0; i < 4; ++i) {
      const float e0 = __expf(c0[i] - mnew); rs += e0; pa[i]      = bfc(e0);
      const float e1 = __expf(c1[i] - mnew); rs += e1; pa[4 + i]  = bfc(e1);
      const float e2 = __expf(c2[i] - mnew); rs += e2; pa[8 + i]  = bfc(e2);
      const float e3 = __expf(c3[i] - mnew); rs += e3; pa[12 + i] = bfc(e3);
    }
    rs += __shfl_xor(rs, 16, 32);
    lrow = lrow * corr + rs;

    // rescale O: element r is row mC+r, whose corr lives in lane mC+r
#pragma unroll
    for (int r = 0; r < 8; ++r) {
      const float cE = __shfl(corr, mC + r, 32);
#pragma unroll
      for (int dt = 0; dt < 4; ++dt) o[dt][r] *= cE;
    }

#pragma unroll
    for (int dt = 0; dt < 4; ++dt)
      o[dt] = wmma_bf16(pa, frag_b(ldsVT[cur], dt * 16, 0, 40, lane), o[dt]);
  }

  // normalize and write ctx (bf16, [B, L, h*dk] row-major)
#pragma unroll
  for (int r = 0; r < 8; ++r) {
    const float lE = __shfl(lrow, mC + r, 32);
    const float inv = 1.0f / lE;
    const int row = rowbase + qb0 + wave * 16 + mC + r;
#pragma unroll
    for (int dt = 0; dt < 4; ++dt)
      ctx[row * kD + hcol + dt * 16 + nC] = f2bf(o[dt][r] * inv);
  }
}

}  // namespace

extern "C" void kernel_launch(void* const* d_in, const int* in_sizes, int n_in,
                              void* d_out, int out_size, void* d_ws, size_t ws_size,
                              hipStream_t stream) {
  (void)in_sizes; (void)n_in; (void)out_size; (void)ws_size;
  // setup_inputs order: word, sent, W1,b1, W2,b2, W3,b3, W4,b4, Wo,bo
  const float* word = (const float*)d_in[0];
  const float* W1 = (const float*)d_in[2];
  const float* b1 = (const float*)d_in[3];
  const float* W2 = (const float*)d_in[4];
  const float* b2 = (const float*)d_in[5];
  // W3/b3 + sent contribute a per-row constant to scores -> softmax-invariant.
  const float* W4 = (const float*)d_in[8];
  const float* b4 = (const float*)d_in[9];
  const float* Wo = (const float*)d_in[10];
  const float* bo = (const float*)d_in[11];

  // workspace (bf16): Q,K,V,ctx [4096x1024] + word_bf16 + 4 weight_bf16
  unsigned short* qb    = (unsigned short*)d_ws;
  unsigned short* kb    = qb + (size_t)kM * kD;
  unsigned short* vb    = kb + (size_t)kM * kD;
  unsigned short* cb    = vb + (size_t)kM * kD;
  unsigned short* wordb = cb + (size_t)kM * kD;
  unsigned short* w1b   = wordb + (size_t)kM * kD;
  unsigned short* w2b   = w1b + (size_t)kD * kD;
  unsigned short* w4b   = w2b + (size_t)kD * kD;
  unsigned short* wob   = w4b + (size_t)kD * kD;

  const int nWord = kM * kD, nW = kD * kD;
  to_bf16_kernel<<<nWord / 2048, 256, 0, stream>>>(word, wordb, nWord);
  to_bf16_kernel<<<nW / 2048, 256, 0, stream>>>(W1, w1b, nW);
  to_bf16_kernel<<<nW / 2048, 256, 0, stream>>>(W2, w2b, nW);
  to_bf16_kernel<<<nW / 2048, 256, 0, stream>>>(W4, w4b, nW);
  to_bf16_kernel<<<nW / 2048, 256, 0, stream>>>(Wo, wob, nW);

  const dim3 gridP(kD / 128, kM / 64);  // (8, 64)
  proj_gemm<true><<<gridP, 256, 0, stream>>>(wordb, w1b, b1, qb);
  proj_gemm<true><<<gridP, 256, 0, stream>>>(wordb, w2b, b2, kb);
  proj_gemm<true><<<gridP, 256, 0, stream>>>(wordb, w4b, b4, vb);

  attn_kernel<<<dim3(kLw / 64, kB * 16), 128, 0, stream>>>(qb, kb, vb, cb);

  proj_gemm<false><<<gridP, 256, 0, stream>>>(cb, wob, bo, (float*)d_out);
}
